// MainBlock_55490977464339
// MI455X (gfx1250) — compile-verified, matrix operand order checked by reading, not run on hardware
//
#include <hip/hip_runtime.h>
#include <hip/hip_bf16.h>
#include <math.h>

// ---------------- constants for this problem ----------------
#define BB   8
#define NN   197
#define DD   768
#define HH   12
#define EE   8
#define HID  3072
#define TT   (BB*NN)        // 1576
#define BH   (BB*HH)        // 96
#define HD   64
#define KPAD 224            // key dim padded to multiple of 32 for WMMA K

typedef __bf16 bf16;
typedef __attribute__((ext_vector_type(16))) __bf16 v16bf;
typedef __attribute__((ext_vector_type(8)))  float  v8f;
typedef __attribute__((ext_vector_type(4)))  unsigned int uint4v;

static __device__ __forceinline__ int imin(int a, int b) { return a < b ? a : b; }

// ---------------- fragment loader (A and B share the same CDNA5 16-bit layout) ----------
// lane L: row = L&15, half h = L>>4 ; elements 0..7 = k in [8h,8h+7], 8..15 = k in [16+8h, 16+8h+7]
static __device__ __forceinline__ v16bf load_frag(const bf16* __restrict__ p, int h) {
    v16bf f;
    uint4v a0 = *(const uint4v*)(p + 8*h);
    uint4v a1 = *(const uint4v*)(p + 16 + 8*h);
    ((uint4v*)&f)[0] = a0;
    ((uint4v*)&f)[1] = a1;
    return f;
}

#define WMMA_BF16(A_, B_, C_) \
    __builtin_amdgcn_wmma_f32_16x16x32_bf16(false, A_, false, B_, (short)0, C_, false, false)

// ---------------- generic bf16 WMMA GEMM: C = A[MxK] * Bt[NxK]^T (+epilogue) ------------
// block tile 64(M) x 256(N); 8 waves: 2 along M x 4 along N; wave tile 32x64.
// Per k-step: 2 A frags x 4 B frags -> 8 WMMAs. Branch-free loads via row clamping:
// out-of-range A rows / B rows are clamped to the last valid row; the garbage lands only
// in D rows/cols that the store-side (mg<M, ng<Nstore) guards skip.
// Software pipeline: (steps-1) iterations load stage s+1 before computing stage s,
// then a drain block computes the final stage with no loads.
// MODE 0: Cf = acc + bias
// MODE 1: Cf = resid + acc + bias
// MODE 2: Cbf = gelu(acc + bias)
// MODE 3: Cf += gate[m*E+eidx] * (acc + bias)
// MODE 4: Cf = acc * scale
// MODE 5: Cbf = acc
template<int MODE>
__global__ __launch_bounds__(256) void gemm_bf16_kernel(
    const bf16* __restrict__ A, int lda, long aBS,
    const bf16* __restrict__ Bt, int ldb, long bBS, int Nb,
    const float* __restrict__ bias,
    float* Cf, bf16* Cbf, int ldc, int Nstore,
    int cdiv, long couter, long cinner,
    int M, int K,
    const float* __restrict__ resid,
    const float* __restrict__ gate, int eidx,
    float scale)
{
    const int z = blockIdx.z;
    A  += (long)z * aBS;
    Bt += (long)z * bBS;
    const long coff = (long)(z / cdiv) * couter + (long)(z % cdiv) * cinner;

    const int lane = threadIdx.x & 31;
    const int wid  = threadIdx.x >> 5;
    const int wm   = wid & 1;        // 0..1  (M)
    const int wn   = wid >> 1;       // 0..3  (N)
    const int h    = lane >> 4;
    const int r    = lane & 15;

    const int m0    = blockIdx.y * 64 + wm * 32;     // wave's first row
    const int nbase = blockIdx.x * 256 + wn * 64;    // wave's first col

    // clamped, loop-invariant row pointers (branch-free inner loop)
    const bf16* pA0 = A + (long)imin(m0 + r,      M - 1) * lda;
    const bf16* pA1 = A + (long)imin(m0 + 16 + r, M - 1) * lda;
    const bf16* pB0 = Bt + (long)imin(nbase +  0 + r, Nb - 1) * ldb;
    const bf16* pB1 = Bt + (long)imin(nbase + 16 + r, Nb - 1) * ldb;
    const bf16* pB2 = Bt + (long)imin(nbase + 32 + r, Nb - 1) * ldb;
    const bf16* pB3 = Bt + (long)imin(nbase + 48 + r, Nb - 1) * ldb;

    v8f acc[8] = {};

    // stage 0 loads
    v16bf a0 = load_frag(pA0, h);
    v16bf a1 = load_frag(pA1, h);
    v16bf b0 = load_frag(pB0, h);
    v16bf b1 = load_frag(pB1, h);
    v16bf b2 = load_frag(pB2, h);
    v16bf b3 = load_frag(pB3, h);

    const int steps = K >> 5;                 // K is always a multiple of 32
    for (int s = 1; s < steps; ++s) {
        pA0 += 32; pA1 += 32;
        pB0 += 32; pB1 += 32; pB2 += 32; pB3 += 32;

        // prefetch one tile ahead toward the WGP caches (global_prefetch_b8)
        __builtin_prefetch(pB0 + 32, 0, 3);
        __builtin_prefetch(pB1 + 32, 0, 3);
        __builtin_prefetch(pB2 + 32, 0, 3);
        __builtin_prefetch(pB3 + 32, 0, 3);
        __builtin_prefetch(pA0 + 32, 0, 3);
        __builtin_prefetch(pA1 + 32, 0, 3);

        // issue next stage's loads before this stage's WMMAs
        v16bf na0 = load_frag(pA0, h);
        v16bf na1 = load_frag(pA1, h);
        v16bf nb0 = load_frag(pB0, h);
        v16bf nb1 = load_frag(pB1, h);
        v16bf nb2 = load_frag(pB2, h);
        v16bf nb3 = load_frag(pB3, h);

        acc[0] = WMMA_BF16(a0, b0, acc[0]);
        acc[4] = WMMA_BF16(a1, b0, acc[4]);
        acc[1] = WMMA_BF16(a0, b1, acc[1]);
        acc[5] = WMMA_BF16(a1, b1, acc[5]);
        acc[2] = WMMA_BF16(a0, b2, acc[2]);
        acc[6] = WMMA_BF16(a1, b2, acc[6]);
        acc[3] = WMMA_BF16(a0, b3, acc[3]);
        acc[7] = WMMA_BF16(a1, b3, acc[7]);

        a0 = na0; a1 = na1;
        b0 = nb0; b1 = nb1; b2 = nb2; b3 = nb3;
    }

    // drain: final stage, no loads
    acc[0] = WMMA_BF16(a0, b0, acc[0]);
    acc[4] = WMMA_BF16(a1, b0, acc[4]);
    acc[1] = WMMA_BF16(a0, b1, acc[1]);
    acc[5] = WMMA_BF16(a1, b1, acc[5]);
    acc[2] = WMMA_BF16(a0, b2, acc[2]);
    acc[6] = WMMA_BF16(a1, b2, acc[6]);
    acc[3] = WMMA_BF16(a0, b3, acc[3]);
    acc[7] = WMMA_BF16(a1, b3, acc[7]);

    // epilogue: D layout -> lane holds column n=lane&15, rows m = i + 8*h
    #pragma unroll
    for (int half = 0; half < 2; ++half) {
        #pragma unroll
        for (int j = 0; j < 4; ++j) {
            const int ng = nbase + j * 16 + r;
            if (ng >= Nstore) continue;
            const float bv = (MODE == 4 || MODE == 5) ? 0.0f : bias[ng];
            #pragma unroll
            for (int i = 0; i < 8; ++i) {
                const int mg = m0 + half * 16 + i + 8 * h;
                if (mg >= M) continue;
                const long ci = coff + (long)mg * ldc + ng;
                const float v = acc[half * 4 + j][i];
                if (MODE == 0)      Cf[ci] = v + bv;
                else if (MODE == 1) Cf[ci] = resid[(long)mg * ldc + ng] + v + bv;
                else if (MODE == 2) {
                    float t = v + bv;
                    Cbf[ci] = (bf16)(0.5f * t * (1.0f + erff(t * 0.70710678f)));
                }
                else if (MODE == 3) Cf[ci] += gate[(long)mg * EE + eidx] * (v + bv);
                else if (MODE == 4) Cf[ci] = v * scale;
                else                Cbf[ci] = (bf16)v;
            }
        }
    }
}

// ---------------- LayerNorm: one block per token, writes bf16 (optionally copies fp32) --
__global__ __launch_bounds__(256) void layernorm_kernel(
    const float* __restrict__ x, const float* __restrict__ w, const float* __restrict__ b,
    bf16* __restrict__ out, float* copyOut)
{
    __shared__ float red[256];
    const int t = blockIdx.x;
    const float* row = x + (long)t * DD;
    const int tid = threadIdx.x;

    float s = 0.f;
    for (int c = tid; c < DD; c += 256) s += row[c];
    red[tid] = s; __syncthreads();
    for (int off = 128; off > 0; off >>= 1) { if (tid < off) red[tid] += red[tid + off]; __syncthreads(); }
    const float mu = red[0] * (1.0f / DD);
    __syncthreads();

    float s2 = 0.f;
    for (int c = tid; c < DD; c += 256) { float d = row[c] - mu; s2 += d * d; }
    red[tid] = s2; __syncthreads();
    for (int off = 128; off > 0; off >>= 1) { if (tid < off) red[tid] += red[tid + off]; __syncthreads(); }
    const float rs = rsqrtf(red[0] * (1.0f / DD) + 1e-5f);

    for (int c = tid; c < DD; c += 256) {
        const float xv = row[c];
        out[(long)t * DD + c] = (bf16)((xv - mu) * rs * w[c] + b[c]);
        if (copyOut) copyOut[(long)t * DD + c] = xv;
    }
}

// ---------------- split QKV (fp32 [T,2304]) into bf16 q,k [BH,N,64] and vT [BH,64,KPAD] -
__global__ void split_qk_kernel(const float* __restrict__ qkv,
                                bf16* __restrict__ qbf, bf16* __restrict__ kbf)
{
    const long i = (long)blockIdx.x * 256 + threadIdx.x;
    if (i >= (long)BH * NN * HD) return;
    const int d  = (int)(i % HD);
    long r = i / HD;
    const int nt = (int)(r % NN); r /= NN;
    const int hh = (int)(r % HH);
    const int bb = (int)(r / HH);
    const long t = (long)bb * NN + nt;
    qbf[i] = (bf16)qkv[t * (3*DD) + 0   + hh * HD + d];
    kbf[i] = (bf16)qkv[t * (3*DD) + DD  + hh * HD + d];
}

__global__ void build_vt_kernel(const float* __restrict__ qkv, bf16* __restrict__ vt)
{
    const long i = (long)blockIdx.x * 256 + threadIdx.x;
    if (i >= (long)BH * HD * KPAD) return;
    const int k  = (int)(i % KPAD);
    long r = i / KPAD;
    const int d  = (int)(r % HD); r /= HD;
    const int hh = (int)(r % HH);
    const int bb = (int)(r / HH);
    float v = 0.f;
    if (k < NN) v = qkv[((long)bb * NN + k) * (3*DD) + 2*DD + hh * HD + d];
    vt[i] = (bf16)v;
}

// ---------------- softmax over key dim: one block per (bh, query-row) -------------------
__global__ __launch_bounds__(256) void softmax_kernel(const float* __restrict__ S, bf16* __restrict__ P)
{
    __shared__ float red[256];
    const long rix = blockIdx.x;                 // over BH*NN rows
    const float* row = S + rix * KPAD;
    bf16* prow = P + rix * KPAD;
    const int tid = threadIdx.x;

    float m = -1e30f;
    for (int c = tid; c < NN; c += 256) m = fmaxf(m, row[c]);
    red[tid] = m; __syncthreads();
    for (int off = 128; off > 0; off >>= 1) { if (tid < off) red[tid] = fmaxf(red[tid], red[tid + off]); __syncthreads(); }
    m = red[0]; __syncthreads();

    float s = 0.f;
    for (int c = tid; c < NN; c += 256) s += __expf(row[c] - m);
    red[tid] = s; __syncthreads();
    for (int off = 128; off > 0; off >>= 1) { if (tid < off) red[tid] += red[tid + off]; __syncthreads(); }
    const float inv = 1.0f / red[0];

    for (int c = tid; c < KPAD; c += 256) {
        const float p = (c < NN) ? __expf(row[c] - m) * inv : 0.0f;
        prow[c] = (bf16)p;
    }
}

// ---------------- gating: one wave per token, top-2 softmax ----------------------------
__global__ __launch_bounds__(256) void gate_kernel(const bf16* __restrict__ xn2,
                                                   const float* __restrict__ wg,
                                                   float* __restrict__ gates)
{
    const int t = blockIdx.x * 8 + (threadIdx.x >> 5);
    const int lane = threadIdx.x & 31;
    if (t >= TT) return;
    float lg[EE];
    #pragma unroll
    for (int e = 0; e < EE; ++e) lg[e] = 0.f;
    for (int k = lane; k < DD; k += 32) {
        const float xv = (float)xn2[(long)t * DD + k];
        #pragma unroll
        for (int e = 0; e < EE; ++e) lg[e] += xv * wg[(long)k * EE + e];
    }
    #pragma unroll
    for (int e = 0; e < EE; ++e)
        for (int off = 16; off > 0; off >>= 1) lg[e] += __shfl_xor(lg[e], off, 32);
    if (lane == 0) {
        int i1 = 0;
        for (int e = 1; e < EE; ++e) if (lg[e] > lg[i1]) i1 = e;
        int i2 = -1;
        for (int e = 0; e < EE; ++e) { if (e == i1) continue; if (i2 < 0 || lg[e] > lg[i2]) i2 = e; }
        const float mx = fmaxf(lg[i1], lg[i2]);
        const float ea = __expf(lg[i1] - mx), eb = __expf(lg[i2] - mx);
        const float inv = 1.0f / (ea + eb);
        float g[EE];
        #pragma unroll
        for (int e = 0; e < EE; ++e) g[e] = 0.f;
        g[i1] = ea * inv; g[i2] = eb * inv;
        for (int e = 0; e < EE; ++e) gates[(long)t * EE + e] = g[e];
    }
}

// ---------------- weight conversion -----------------------------------------------------
__global__ void cvt_kernel(const float* __restrict__ in, bf16* __restrict__ out, long n)
{
    const long i = (long)blockIdx.x * 256 + threadIdx.x;
    if (i < n) out[i] = (bf16)in[i];
}

// out[e][n][k] = in[e][k][n]  (make weights N-major for the B fragment loader)
__global__ void transpose_cvt_kernel(const float* __restrict__ in, bf16* __restrict__ out,
                                     int Edim, int Kdim, int Ndim)
{
    const long per = (long)Kdim * Ndim;
    const long i = (long)blockIdx.x * 256 + threadIdx.x;
    if (i >= (long)Edim * per) return;
    const int e = (int)(i / per);
    const long rem = i % per;
    const int n = (int)(rem / Kdim);
    const int k = (int)(rem % Kdim);
    out[i] = (bf16)in[(long)e * per + (long)k * Ndim + n];
}

// ---------------- host side -------------------------------------------------------------
static inline dim3 gemm_grid(int M, int Ncols, int z) {
    return dim3((Ncols + 255) / 256, (M + 63) / 64, z);
}

extern "C" void kernel_launch(void* const* d_in, const int* in_sizes, int n_in,
                              void* d_out, int out_size, void* d_ws, size_t ws_size,
                              hipStream_t stream) {
    const float* x       = (const float*)d_in[0];
    const float* n1w     = (const float*)d_in[1];
    const float* n1b     = (const float*)d_in[2];
    const float* qkv_w   = (const float*)d_in[3];
    const float* qkv_b   = (const float*)d_in[4];
    const float* proj_w  = (const float*)d_in[5];
    const float* proj_b  = (const float*)d_in[6];
    const float* n2w     = (const float*)d_in[7];
    const float* n2b     = (const float*)d_in[8];
    const float* w_gate  = (const float*)d_in[9];
    const float* W1      = (const float*)d_in[10];
    const float* b1      = (const float*)d_in[11];
    const float* W2      = (const float*)d_in[12];
    const float* b2      = (const float*)d_in[13];
    float* out = (float*)d_out;

    // workspace carve-up (256B aligned)
    char* w = (char*)d_ws;
    size_t cur = 0;
    auto alloc = [&](size_t sz) { void* p = w + cur; cur = (cur + sz + 255) & ~(size_t)255; return p; };
    bf16*  qkvw_bf = (bf16*)alloc((size_t)3*DD*DD*2);
    bf16*  projw_bf= (bf16*)alloc((size_t)DD*DD*2);
    bf16*  W1t     = (bf16*)alloc((size_t)EE*HID*DD*2);
    bf16*  W2t     = (bf16*)alloc((size_t)EE*DD*HID*2);
    bf16*  xnbf    = (bf16*)alloc((size_t)TT*DD*2);
    float* qkvf    = (float*)alloc((size_t)TT*3*DD*4);
    bf16*  qbf     = (bf16*)alloc((size_t)BH*NN*HD*2);
    bf16*  kbf     = (bf16*)alloc((size_t)BH*NN*HD*2);
    bf16*  vtbf    = (bf16*)alloc((size_t)BH*HD*KPAD*2);
    float* S       = (float*)alloc((size_t)BH*NN*KPAD*4);
    bf16*  Pbf     = (bf16*)alloc((size_t)BH*NN*KPAD*2);
    bf16*  Obf     = (bf16*)alloc((size_t)TT*DD*2);
    float* x1      = (float*)alloc((size_t)TT*DD*4);
    bf16*  xn2bf   = (bf16*)alloc((size_t)TT*DD*2);
    float* gates   = (float*)alloc((size_t)TT*EE*4);
    bf16*  Hbf     = (bf16*)alloc((size_t)TT*HID*2);
    (void)ws_size; (void)in_sizes; (void)n_in; (void)out_size;

    // 1) convert weights to bf16 (N-major)
    { long n = (long)3*DD*DD; cvt_kernel<<<(int)((n+255)/256),256,0,stream>>>(qkv_w, qkvw_bf, n); }
    { long n = (long)DD*DD;   cvt_kernel<<<(int)((n+255)/256),256,0,stream>>>(proj_w, projw_bf, n); }
    { long n = (long)EE*DD*HID; transpose_cvt_kernel<<<(int)((n+255)/256),256,0,stream>>>(W1, W1t, EE, DD, HID); }
    { long n = (long)EE*HID*DD; transpose_cvt_kernel<<<(int)((n+255)/256),256,0,stream>>>(W2, W2t, EE, HID, DD); }

    // 2) LN1 -> xn (bf16)
    layernorm_kernel<<<TT, 256, 0, stream>>>(x, n1w, n1b, xnbf, nullptr);

    // 3) qkv = xn @ qkv_w^T + b  -> fp32 [T, 2304]
    gemm_bf16_kernel<0><<<gemm_grid(TT, 3*DD, 1), 256, 0, stream>>>(
        xnbf, DD, 0, qkvw_bf, DD, 0, 3*DD, qkv_b,
        qkvf, nullptr, 3*DD, 3*DD, 1, 0, 0, TT, DD, nullptr, nullptr, 0, 0.f);

    // 4) split q,k (bf16) and v^T padded (bf16)
    { long n = (long)BH*NN*HD;   split_qk_kernel<<<(int)((n+255)/256),256,0,stream>>>(qkvf, qbf, kbf); }
    { long n = (long)BH*HD*KPAD; build_vt_kernel<<<(int)((n+255)/256),256,0,stream>>>(qkvf, vtbf); }

    // 5) scores S = (q @ k^T) * hd^-0.5   per (b,h)
    gemm_bf16_kernel<4><<<gemm_grid(NN, NN, BH), 256, 0, stream>>>(
        qbf, HD, (long)NN*HD, kbf, HD, (long)NN*HD, NN, nullptr,
        S, nullptr, KPAD, KPAD, 1, (long)NN*KPAD, 0, NN, HD, nullptr, nullptr, 0, 0.125f);

    // 6) softmax rows -> Pbf (zero-padded K to 224)
    softmax_kernel<<<BH*NN, 256, 0, stream>>>(S, Pbf);

    // 7) O = P @ V, stored directly as [B,N,H*hd] bf16
    gemm_bf16_kernel<5><<<gemm_grid(NN, HD, BH), 256, 0, stream>>>(
        Pbf, KPAD, (long)NN*KPAD, vtbf, KPAD, (long)HD*KPAD, HD, nullptr,
        nullptr, Obf, DD, HD, HH, (long)NN*DD, (long)HD, NN, KPAD, nullptr, nullptr, 0, 0.f);

    // 8) x1 = x + O @ proj_w^T + proj_b   (fp32)
    gemm_bf16_kernel<1><<<gemm_grid(TT, DD, 1), 256, 0, stream>>>(
        Obf, DD, 0, projw_bf, DD, 0, DD, proj_b,
        x1, nullptr, DD, DD, 1, 0, 0, TT, DD, x, nullptr, 0, 0.f);

    // 9) LN2 -> xn2 (bf16), and seed d_out = x1
    layernorm_kernel<<<TT, 256, 0, stream>>>(x1, n2w, n2b, xn2bf, out);

    // 10) gating: top-2 softmax -> gates [T,E] (zeros elsewhere)
    gate_kernel<<<(TT + 7) / 8, 256, 0, stream>>>(xn2bf, w_gate, gates);

    // 11) MoE: dense per-expert, gate-weighted accumulate into d_out
    for (int e = 0; e < EE; ++e) {
        gemm_bf16_kernel<2><<<gemm_grid(TT, HID, 1), 256, 0, stream>>>(
            xn2bf, DD, 0, W1t + (long)e*HID*DD, DD, 0, HID, b1 + (long)e*HID,
            nullptr, Hbf, HID, HID, 1, 0, 0, TT, DD, nullptr, nullptr, 0, 0.f);
        gemm_bf16_kernel<3><<<gemm_grid(TT, DD, 1), 256, 0, stream>>>(
            Hbf, HID, 0, W2t + (long)e*DD*HID, HID, 0, DD, b2 + (long)e*DD,
            out, nullptr, DD, DD, 1, 0, 0, TT, HID, nullptr, gates, e, 0.f);
    }
}
